// AttentionPooling_67302137528446
// MI455X (gfx1250) — matrix operand order repaired and checked by reading, not compiled, not run on hardware
//
#include <hip/hip_runtime.h>
#include <hip/hip_bf16.h>

typedef __attribute__((ext_vector_type(16))) __bf16 v16bf;
typedef __attribute__((ext_vector_type(8)))  __bf16 v8bf;
typedef __attribute__((ext_vector_type(4)))  __bf16 v4bf;
typedef __attribute__((ext_vector_type(8)))  float  v8f;
typedef __attribute__((ext_vector_type(4)))  unsigned int u32x4;
typedef __attribute__((ext_vector_type(8)))  int i32x8;
typedef __attribute__((ext_vector_type(4)))  int i32x4;

#define C_DIM 512
#define KPTS  16

__device__ __forceinline__ v16bf pack16(float4 q0, float4 q1, float4 q2, float4 q3) {
    v16bf r;
    r[0]  = (__bf16)q0.x; r[1]  = (__bf16)q0.y; r[2]  = (__bf16)q0.z; r[3]  = (__bf16)q0.w;
    r[4]  = (__bf16)q1.x; r[5]  = (__bf16)q1.y; r[6]  = (__bf16)q1.z; r[7]  = (__bf16)q1.w;
    r[8]  = (__bf16)q2.x; r[9]  = (__bf16)q2.y; r[10] = (__bf16)q2.z; r[11] = (__bf16)q2.w;
    r[12] = (__bf16)q3.x; r[13] = (__bf16)q3.y; r[14] = (__bf16)q3.z; r[15] = (__bf16)q3.w;
    return r;
}

// Prologue: one-time f32 -> bf16 conversion of both weight matrices into d_ws.
__global__ __launch_bounds__(256) void cvt_weights(
    const float* __restrict__ ws_f, const float* __restrict__ wc_f,
    __bf16* __restrict__ ws_b, __bf16* __restrict__ wc_b)
{
    const int i = (blockIdx.x * 256 + threadIdx.x) * 4;
    float4 fa = *(const float4*)(ws_f + i);
    float4 fb = *(const float4*)(wc_f + i);
    v4bf ra, rb;
    ra[0] = (__bf16)fa.x; ra[1] = (__bf16)fa.y; ra[2] = (__bf16)fa.z; ra[3] = (__bf16)fa.w;
    rb[0] = (__bf16)fb.x; rb[1] = (__bf16)fb.y; rb[2] = (__bf16)fb.z; rb[3] = (__bf16)fb.w;
    *(v4bf*)(ws_b + i) = ra;
    *(v4bf*)(wc_b + i) = rb;
}

// Stage A: per-token logits GEMM (bf16 WMMA, f32 acc), k-softmax, pooling.
// One block (4 waves) per token; wave w owns columns [w*128, w*128+128).
// x tile is DMA'd into LDS by the Tensor Data Mover (one descriptor, wave 0).
__global__ __launch_bounds__(128) void attn_pool_stageA(
    const float* __restrict__ x, const __bf16* __restrict__ w_score_b,
    __bf16* __restrict__ pooled)
{
    __shared__ float xs[KPTS * C_DIM];  // 32 KB token tile (k-major, c contiguous)

    const int tid   = threadIdx.x;
    const int token = blockIdx.x;

    // Real LDS byte offset of xs (low 32 bits of the generic pointer).
    // The ptrtoint also escapes xs so the TDM write can't be DCE'd away.
    const unsigned lds_off = (unsigned)(size_t)(&xs[0]);

    // --- TDM: DMA the 16x512 f32 tile global -> LDS ----------------------------
    if (tid < 32) {  // wave 0 only; TDM is wave-level (EXEC ignored)
        unsigned long long gaddr =
            (unsigned long long)(x + (size_t)token * (KPTS * C_DIM));
        u32x4 g0;
        g0[0] = 1u;                                  // count=1, user mode
        g0[1] = lds_off;                             // lds_addr (bytes)
        g0[2] = (unsigned int)(gaddr & 0xffffffffu); // global_addr[31:0]
        g0[3] = (unsigned int)((gaddr >> 32) & 0x1ffffffu) | (2u << 30); // addr[56:32], type=2

        i32x8 g1;
        g1[0] = 0x00020000;        // workgroup_mask=0, data_size=2 (4 bytes), no pad
        g1[1] = (C_DIM << 16);     // tensor_dim0[15:0] = 512 (bits 63:48)
        g1[2] = (KPTS << 16);      // tensor_dim0[31:16]=0, tensor_dim1[15:0] = 16
        g1[3] = (C_DIM << 16);     // tensor_dim1[31:16]=0, tile_dim0 = 512
        g1[4] = KPTS;              // tile_dim1 = 16, tile_dim2 = 0
        g1[5] = C_DIM;             // tensor_dim0_stride[31:0] = 512 elements
        g1[6] = 0;                 // stride hi / tensor_dim1_stride lo (unused, 2D)
        g1[7] = 0;

        i32x4 z4 = {0, 0, 0, 0};             // groups 2/3 unused for 2D tile
        i32x8 z8 = {0, 0, 0, 0, 0, 0, 0, 0}; // 6-arg form: extra group (unused)
        __builtin_amdgcn_tensor_load_to_lds(g0, g1, z4, z4, z8, 0);
        __builtin_amdgcn_s_wait_tensorcnt(0);
    }
    asm volatile("" ::: "memory");  // compiler barrier: LDS now holds the tile
    __syncthreads();
    // ---------------------------------------------------------------------------

    const int lane = tid & 31;
    const int h    = lane >> 4;      // wave half (C-layout: M offset 8*h)
    const int ml   = lane & 15;      // row/col within half
    const int wave = tid >> 5;
    const int dbase = wave * 128;

    v8f acc[8];
#pragma unroll
    for (int j = 0; j < 8; ++j)
        acc[j] = (v8f){0.f, 0.f, 0.f, 0.f, 0.f, 0.f, 0.f, 0.f};

    for (int k0 = 0; k0 < C_DIM; k0 += 32) {
        // A fragment (16x32 bf16): low lanes row=ml K={k0..k0+7, k0+16..k0+23},
        // high lanes K shifted by 8 — contiguous 32B chunks in LDS.
        const float* xr = xs + ml * C_DIM + k0 + 8 * h;
        float4 f0 = *(const float4*)(xr);
        float4 f1 = *(const float4*)(xr + 4);
        float4 f2 = *(const float4*)(xr + 16);
        float4 f3 = *(const float4*)(xr + 20);
        v16bf a = pack16(f0, f1, f2, f3);

#pragma unroll
        for (int j = 0; j < 8; ++j) {
            // B fragment (32x16 bf16): lane = output column d, K as in A (8-K runs at 8h, 16+8h)
            // (16B chunks, pre-converted bf16 -> two b128 loads, zero cvt VALU).
            const __bf16* bp = w_score_b + (size_t)(dbase + j * 16 + ml) * C_DIM + k0 + 8 * h;
            v8bf blo = *(const v8bf*)(bp);
            v8bf bhi = *(const v8bf*)(bp + 16);
            v16bf b;
#pragma unroll
            for (int i = 0; i < 8; ++i) { b[i] = blo[i]; b[8 + i] = bhi[i]; }
            acc[j] = __builtin_amdgcn_wmma_f32_16x16x32_bf16(
                false, a, false, b, (short)0, acc[j], false, false);
        }
    }

    // Softmax over k (M dim of the C tile) + pooling.
    // C layout: lane half owns M = r + 8*h for VGPR r; halves exchanged via shfl_xor(16).
#pragma unroll
    for (int j = 0; j < 8; ++j) {
        v8f c = acc[j];
        float m = c[0];
#pragma unroll
        for (int r = 1; r < 8; ++r) m = fmaxf(m, c[r]);
        m = fmaxf(m, __shfl_xor(m, 16, 32));

        float e[8];
        float s = 0.f;
#pragma unroll
        for (int r = 0; r < 8; ++r) {
            e[r] = exp2f((c[r] - m) * 1.4426950408889634f);
            s += e[r];
        }
        s += __shfl_xor(s, 16, 32);
        const float inv = 1.0f / s;

        const int col = dbase + j * 16 + ml;
        float p = 0.f;
#pragma unroll
        for (int r = 0; r < 8; ++r)
            p += xs[(r + 8 * h) * C_DIM + col] * (e[r] * inv);
        p += __shfl_xor(p, 16, 32);

        if (h == 0)
            pooled[(size_t)token * C_DIM + col] = (__bf16)p;
    }
}

// Stage B: y = pooled @ w_conv^T (bf16 WMMA over 16-token M tiles), fused BN + ReLU.
// One block (4 waves) per 16 tokens; wave w owns channels [w*128, w*128+128).
__global__ __launch_bounds__(128) void attn_pool_stageB(
    const __bf16* __restrict__ pooled, const __bf16* __restrict__ w_conv_b,
    const float* __restrict__ gamma, const float* __restrict__ beta,
    const float* __restrict__ mean, const float* __restrict__ var,
    float* __restrict__ out)
{
    const int tid  = threadIdx.x;
    const int lane = tid & 31;
    const int h    = lane >> 4;
    const int ml   = lane & 15;
    const int wave = tid >> 5;
    const int t0    = blockIdx.x * 16;
    const int cbase = wave * 128;

    v8f acc[8];
#pragma unroll
    for (int j = 0; j < 8; ++j)
        acc[j] = (v8f){0.f, 0.f, 0.f, 0.f, 0.f, 0.f, 0.f, 0.f};

    for (int k0 = 0; k0 < C_DIM; k0 += 32) {
        // A fragment from bf16 pooled: row = token t0+ml, two 16B chunks.
        const __bf16* pr = pooled + (size_t)(t0 + ml) * C_DIM + k0 + 8 * h;
        v8bf lo = *(const v8bf*)(pr);
        v8bf hi = *(const v8bf*)(pr + 16);
        v16bf a;
#pragma unroll
        for (int i = 0; i < 8; ++i) { a[i] = lo[i]; a[8 + i] = hi[i]; }

#pragma unroll
        for (int j = 0; j < 8; ++j) {
            const __bf16* bp = w_conv_b + (size_t)(cbase + j * 16 + ml) * C_DIM + k0 + 8 * h;
            v8bf blo = *(const v8bf*)(bp);
            v8bf bhi = *(const v8bf*)(bp + 16);
            v16bf b;
#pragma unroll
            for (int i = 0; i < 8; ++i) { b[i] = blo[i]; b[8 + i] = bhi[i]; }
            acc[j] = __builtin_amdgcn_wmma_f32_16x16x32_bf16(
                false, a, false, b, (short)0, acc[j], false, false);
        }
    }

    // Fused eval-BN + ReLU epilogue; C layout: lane = channel, VGPR r -> token r+8*h.
#pragma unroll
    for (int j = 0; j < 8; ++j) {
        const int c   = cbase + j * 16 + ml;
        const float s = gamma[c] * rsqrtf(var[c] + 1e-5f);
        const float d = beta[c] - mean[c] * s;
#pragma unroll
        for (int r = 0; r < 8; ++r) {
            float y = fmaxf(acc[j][r] * s + d, 0.f);
            out[(size_t)(t0 + r + 8 * h) * C_DIM + c] = y;
        }
    }
}

extern "C" void kernel_launch(void* const* d_in, const int* in_sizes, int n_in,
                              void* d_out, int out_size, void* d_ws, size_t ws_size,
                              hipStream_t stream) {
    const float* x       = (const float*)d_in[0];
    const float* w_score = (const float*)d_in[1];
    const float* w_conv  = (const float*)d_in[2];
    const float* g       = (const float*)d_in[3];
    const float* be      = (const float*)d_in[4];
    const float* mu      = (const float*)d_in[5];
    const float* va      = (const float*)d_in[6];
    float* out = (float*)d_out;

    const int tokens = in_sizes[0] / (KPTS * C_DIM);  // B*N = 16384
    const int welems = C_DIM * C_DIM;                 // 262144 per weight matrix

    // Workspace layout: [pooled bf16: tokens*512][w_score bf16: 256K][w_conv bf16: 256K]
    __bf16* pooled = (__bf16*)d_ws;
    __bf16* wsb    = pooled + (size_t)tokens * C_DIM;
    __bf16* wcb    = wsb + welems;

    cvt_weights<<<welems / (256 * 4), 256, 0, stream>>>(w_score, w_conv, wsb, wcb);
    attn_pool_stageA<<<tokens, 128, 0, stream>>>(x, wsb, pooled);
    attn_pool_stageB<<<tokens / 16, 128, 0, stream>>>(pooled, wcb, g, be, mu, va, out);
}
